// TMLinKernelAttn_28750511079704
// MI455X (gfx1250) — matrix operand order repaired and checked by reading, not compiled, not run on hardware
//
#include <hip/hip_runtime.h>
#include <math.h>

// ---------------------------------------------------------------------------
// Types
// ---------------------------------------------------------------------------
typedef __attribute__((ext_vector_type(4)))  unsigned int u32x4;
typedef __attribute__((ext_vector_type(8)))  float        v8f;
typedef __attribute__((ext_vector_type(16))) __bf16       v16bf;

union Frag  { u32x4 u[2]; v16bf v; };                 // 32 bytes: one wmma bf16 operand
union Ld8   { u32x4 q;    unsigned short s[8];  };

static __device__ __forceinline__ v8f vzero8() {
    v8f r = {0.f,0.f,0.f,0.f,0.f,0.f,0.f,0.f};
    return r;
}

static __device__ __forceinline__ unsigned short f2bf(float f) {
    unsigned u = __float_as_uint(f);
    unsigned r = u + 0x7FFFu + ((u >> 16) & 1u);      // round-to-nearest-even
    return (unsigned short)(r >> 16);
}

static __device__ __forceinline__ v8f wmma_bf16(const Frag& a, const Frag& b, v8f c) {
    return __builtin_amdgcn_wmma_f32_16x16x32_bf16(
        /*neg_a=*/false, a.v, /*neg_b=*/false, b.v,
        /*c_mod=*/(short)0, c, /*reuse_a=*/false, /*reuse_b=*/false);
}

// Problem constants
#define BB   4
#define SS   8192
#define EE   1024
#define HH   16
#define FF   1024
#define FH   64          // per-head feature dim
#define DVV  64
#define TIN  128
#define MROWS (BB*SS)    // 32768

// ---------------------------------------------------------------------------
// fp32 -> bf16 conversion (elementwise)
// ---------------------------------------------------------------------------
__global__ void cvt_bf16_kernel(const float* __restrict__ in,
                                unsigned short* __restrict__ out, size_t n) {
    size_t i = (size_t)blockIdx.x * blockDim.x + threadIdx.x;
    size_t stride = (size_t)gridDim.x * blockDim.x;
    for (; i < n; i += stride) out[i] = f2bf(in[i]);
}

// fp32 (K,N) -> bf16 transposed (N,K), tiled 32x32 through LDS
__global__ __launch_bounds__(256) void transpose_cvt_kernel(
    const float* __restrict__ in, unsigned short* __restrict__ outT,
    int K, int N) {
    __shared__ unsigned short s[32][33];
    int k0 = blockIdx.x * 32, n0 = blockIdx.y * 32;
    int tc = threadIdx.x & 31, tr = threadIdx.x >> 5;   // 32 cols x 8 rows
#pragma unroll
    for (int i = 0; i < 32; i += 8)
        s[tr + i][tc] = f2bf(in[(size_t)(k0 + tr + i) * N + n0 + tc]);
    __syncthreads();
#pragma unroll
    for (int i = 0; i < 32; i += 8)
        outT[(size_t)(n0 + tr + i) * K + k0 + tc] = s[tc][tr + i];
}

__global__ void zero_f32_kernel(float* __restrict__ p, size_t n) {
    size_t i = (size_t)blockIdx.x * blockDim.x + threadIdx.x;
    size_t stride = (size_t)gridDim.x * blockDim.x;
    for (; i < n; i += stride) p[i] = 0.f;
}

// ---------------------------------------------------------------------------
// T = softplus(temp_input @ Wt + bt)   (B*S,128)@(128,16) -> (B*S,16)
// ---------------------------------------------------------------------------
__global__ __launch_bounds__(256) void temp_kernel(
    const float* __restrict__ ti, const float* __restrict__ Wt,
    const float* __restrict__ bt, float* __restrict__ T) {
    __shared__ float sW[TIN * HH];      // 8 KB
    __shared__ float sR[16][TIN];       // 8 KB
    int tid = threadIdx.x;
    int r0  = blockIdx.x * 16;
    for (int t = tid; t < TIN * HH; t += 256) sW[t] = Wt[t];
    for (int t = tid; t < 16 * TIN; t += 256) {
        int rr = t >> 7, cc = t & 127;
        sR[rr][cc] = ti[(size_t)(r0 + rr) * TIN + cc];
    }
    __syncthreads();
    int rr = tid >> 4, h = tid & 15;
    float s = bt[h];
#pragma unroll 8
    for (int c = 0; c < TIN; c++) s += sR[rr][c] * sW[c * HH + h];
    float sp = (s > 20.f) ? s : log1pf(__expf(s));
    T[(size_t)(r0 + rr) * HH + h] = sp;
}

// ---------------------------------------------------------------------------
// Big GEMM: C[M,N] = act(A[M,K] @ W[K,N] + bias), W given PRE-TRANSPOSED (N,K)
//   bf16 in, bf16 out, f32 WMMA accumulation.
//   act 0: none    act 1: elu(x)+1    act 2: (elu(x)+1) * T[row, col>>6]
// 128x128 block tile, 8 waves (2Mx4N), K-step 64 (2 wmma sub-steps),
// double-buffered LDS staged with GLOBAL_LOAD_ASYNC_TO_LDS_B128 (ASYNCcnt).
// ---------------------------------------------------------------------------
#define LDT 72           // padded stride (halves) -> 144B rows keep 16B alignment

__global__ __launch_bounds__(256) void gemm_bias_act_kernel(
    const unsigned short* __restrict__ A,    // (M,K) bf16
    const unsigned short* __restrict__ WT,   // (N,K) bf16 (transposed weights)
    const float* __restrict__ bias,
    const float* __restrict__ T,
    unsigned short* __restrict__ C,
    int M, int N, int K, int act) {
    __shared__ alignas(16) unsigned short As[2][128 * LDT];  // 36 KB
    __shared__ alignas(16) unsigned short Bs[2][128 * LDT];  // 36 KB ([col][k])

    const int tid  = threadIdx.x;
    const int lane = tid & 31;
    const int wave = tid >> 5;
    const int bn   = blockIdx.x * 128;
    const int bm   = blockIdx.y * 128;
    const int wm   = (wave >> 2) * 64;   // wave M offset
    const int wn   = (wave & 3) * 32;    // wave N offset

    // cooperative staging: thread -> (row 0..127, 32-half segment 0/32)
    const int ar   = tid >> 1;
    const int aseg = (tid & 1) * 32;

    v8f acc[4][2];
#pragma unroll
    for (int i = 0; i < 4; i++)
#pragma unroll
        for (int j = 0; j < 2; j++) acc[i][j] = vzero8();

    auto stage = [&](int kt, int buf) {
        // A tile: 128 x 64, 64 contiguous bytes per thread, direct to LDS
        unsigned ldsA = (unsigned)(uintptr_t)&As[buf][ar * LDT + aseg];
        const unsigned short* gA = &A[(size_t)(bm + ar) * K + kt + aseg];
        asm volatile("global_load_async_to_lds_b128 %0, %1, off\n\t"
                     "global_load_async_to_lds_b128 %0, %1, off offset:32"
                     :: "v"(ldsA), "v"(gA) : "memory");
        // B tile from transposed weights: [col][k], same contiguous pattern
        unsigned ldsB = (unsigned)(uintptr_t)&Bs[buf][ar * LDT + aseg];
        const unsigned short* gB = &WT[(size_t)(bn + ar) * K + kt + aseg];
        asm volatile("global_load_async_to_lds_b128 %0, %1, off\n\t"
                     "global_load_async_to_lds_b128 %0, %1, off offset:32"
                     :: "v"(ldsB), "v"(gB) : "memory");
    };

    stage(0, 0);
    asm volatile("s_wait_asynccnt 0x0" ::: "memory");
    __syncthreads();

    const int r   = lane & 15;
    const int kbA = (lane & 16) ? 8 : 0;
    const int kbB = (lane & 16) ? 8 : 0;

    for (int kt = 0; kt < K; kt += 64) {
        int buf = (kt >> 6) & 1;
        if (kt + 64 < K) stage(kt + 64, buf ^ 1);

#pragma unroll
        for (int ks = 0; ks < 64; ks += 32) {
            Frag afr[4], bfr[2];
#pragma unroll
            for (int i = 0; i < 4; i++) {
                int row = wm + i * 16 + r;
                afr[i].u[0] = *(const u32x4*)&As[buf][row * LDT + ks + kbA];
                afr[i].u[1] = *(const u32x4*)&As[buf][row * LDT + ks + 16 + kbA];
            }
#pragma unroll
            for (int j = 0; j < 2; j++) {
                int col = wn + j * 16 + r;
                bfr[j].u[0] = *(const u32x4*)&Bs[buf][col * LDT + ks + kbB];
                bfr[j].u[1] = *(const u32x4*)&Bs[buf][col * LDT + ks + kbB + 16];
            }
#pragma unroll
            for (int i = 0; i < 4; i++)
#pragma unroll
                for (int j = 0; j < 2; j++)
                    acc[i][j] = wmma_bf16(afr[i], bfr[j], acc[i][j]);
        }
        asm volatile("s_wait_asynccnt 0x0" ::: "memory");
        __syncthreads();
    }

    // epilogue
#pragma unroll
    for (int i = 0; i < 4; i++)
#pragma unroll
        for (int j = 0; j < 2; j++) {
            int cn = bn + wn + j * 16 + (lane & 15);
            float bs = bias ? bias[cn] : 0.f;
#pragma unroll
            for (int vv = 0; vv < 8; vv++) {
                int rm = bm + wm + i * 16 + vv + ((lane & 16) ? 8 : 0);
                float val = acc[i][j][vv] + bs;
                if (act >= 1) val = (val > 0.f) ? (val + 1.f) : __expf(val);
                if (act == 2) val *= T[(size_t)rm * HH + (cn >> 6)];
                C[(size_t)rm * N + cn] = f2bf(val);
            }
        }
}

// ---------------------------------------------------------------------------
// kv kernel: per (b,h):  kv_aug[f, n] = sum_s pk[s,f] * vaug[s,n]
//   vaug cols: 0..63 = v, 64 = 1.0 (gives z), 65..95 = 0 (pad)
//   grid (B*H, 16 k-slices), f32 atomicAdd into zeroed kvf (B*H,64,80)
// ---------------------------------------------------------------------------
#define KSLICES 16

__global__ __launch_bounds__(256) void kv_kernel(
    const unsigned short* __restrict__ pk,
    const unsigned short* __restrict__ v,
    float* __restrict__ kvf) {
    __shared__ alignas(16) unsigned short AT[64 * 40];  // pk transposed [f][k]
    __shared__ alignas(16) unsigned short BT[96 * 40];  // vaug transposed [n][k]

    const int tid  = threadIdx.x;
    const int lane = tid & 31;
    const int wave = tid >> 5;
    const int bh   = blockIdx.x;
    const int b    = bh >> 4, h = bh & 15;
    const int s0   = blockIdx.y * (SS / KSLICES);       // 512-row slice

    const int wf  = (wave >> 1) * 16;                   // f-tile offset (0..48)
    const int wn0 = (wave & 1) * 48;                    // n offset: 3 tiles each

    // constant augmented rows (64 = ones, 65..95 = zeros), written once
    for (int t = tid; t < 32 * 32; t += 256) {
        int n = 64 + (t >> 5), k = t & 31;
        BT[n * 40 + k] = (n == 64) ? (unsigned short)0x3F80 : (unsigned short)0;
    }

    v8f acc[3];
#pragma unroll
    for (int j = 0; j < 3; j++) acc[j] = vzero8();

    const int sr = tid >> 3;          // staging row 0..31
    const int f0 = (tid & 7) * 8;     // 8 contiguous halves
    const int r   = lane & 15;
    const int kbA = (lane & 16) ? 8 : 0;
    const int kbB = (lane & 16) ? 8 : 0;

    for (int ks = 0; ks < SS / KSLICES; ks += 32) {
        __syncthreads();
        size_t rowOff = ((size_t)(b * SS + s0 + ks + sr)) * FF + h * FH + f0;
        Ld8 la, lv;
        la.q = *(const u32x4*)&pk[rowOff];
        lv.q = *(const u32x4*)&v[rowOff];
#pragma unroll
        for (int i = 0; i < 8; i++) {
            AT[(f0 + i) * 40 + sr] = la.s[i];
            BT[(f0 + i) * 40 + sr] = lv.s[i];
        }
        __syncthreads();

        Frag a;
        a.u[0] = *(const u32x4*)&AT[(wf + r) * 40 + kbA];
        a.u[1] = *(const u32x4*)&AT[(wf + r) * 40 + 16 + kbA];
#pragma unroll
        for (int j = 0; j < 3; j++) {
            Frag bb;
            int n = wn0 + j * 16 + r;
            bb.u[0] = *(const u32x4*)&BT[n * 40 + kbB];
            bb.u[1] = *(const u32x4*)&BT[n * 40 + kbB + 16];
            acc[j] = wmma_bf16(a, bb, acc[j]);
        }
    }

#pragma unroll
    for (int j = 0; j < 3; j++) {
        int n = wn0 + j * 16 + (lane & 15);
        if (n < 80) {
#pragma unroll
            for (int vv = 0; vv < 8; vv++) {
                int f = wf + vv + ((lane & 16) ? 8 : 0);
                atomicAdd(&kvf[((size_t)bh * 64 + f) * 80 + n], acc[j][vv]);
            }
        }
    }
}

// kvf (B*H,64,80) f32 -> kvT (B*H,96,64) bf16 transposed, rows 80..95 zero
__global__ void cvt_kv_kernel(const float* __restrict__ kvf,
                              unsigned short* __restrict__ kvT) {
    size_t ntot = (size_t)BB * HH * 96 * 64;
    size_t i = (size_t)blockIdx.x * blockDim.x + threadIdx.x;
    size_t stride = (size_t)gridDim.x * blockDim.x;
    for (; i < ntot; i += stride) {
        int bh = (int)(i / (96 * 64));
        int rem = (int)(i % (96 * 64));
        int n = rem / 64, f = rem % 64;
        float val = (n < 80) ? kvf[((size_t)bh * 64 + f) * 80 + n] : 0.f;
        kvT[i] = f2bf(val);
    }
}

// ---------------------------------------------------------------------------
// Output: per (b,h, 128-row slab): phi_q(128x64) @ kv_aug(64x96) via WMMA,
// col 64 = den; divide and store fp32.
// ---------------------------------------------------------------------------
__global__ __launch_bounds__(256) void out_kernel(
    const unsigned short* __restrict__ phiq,
    const unsigned short* __restrict__ kvT,   // (B*H, 96, 64) [n][f]
    float* __restrict__ out) {
    __shared__ alignas(16) unsigned short BT[96 * 72];  // [n][f] pad->72
    __shared__ float den_s[128];

    const int tid  = threadIdx.x;
    const int lane = tid & 31;
    const int wave = tid >> 5;
    const int bh   = blockIdx.x;
    const int b    = bh >> 4, h = bh & 15;
    const int s0   = blockIdx.y * 128;

    for (int t = tid; t < 96 * 8; t += 256) {
        int n = t >> 3, seg = (t & 7) * 8;
        *(u32x4*)&BT[n * 72 + seg] = *(const u32x4*)&kvT[((size_t)bh * 96 + n) * 64 + seg];
    }
    __syncthreads();

    v8f acc[6];
#pragma unroll
    for (int j = 0; j < 6; j++) acc[j] = vzero8();

    const int r   = lane & 15;
    const int kbA = (lane & 16) ? 8 : 0;
    const int kbB = (lane & 16) ? 8 : 0;

#pragma unroll
    for (int kt = 0; kt < 64; kt += 32) {
        Frag a;
        const unsigned short* ap =
            &phiq[((size_t)(b * SS + s0 + wave * 16 + r)) * FF + h * FH + kt];
        a.u[0] = *(const u32x4*)(ap + kbA);
        a.u[1] = *(const u32x4*)(ap + 16 + kbA);
#pragma unroll
        for (int j = 0; j < 6; j++) {
            Frag bb;
            int n = j * 16 + r;
            bb.u[0] = *(const u32x4*)&BT[n * 72 + kt + kbB];
            bb.u[1] = *(const u32x4*)&BT[n * 72 + kt + kbB + 16];
            acc[j] = wmma_bf16(a, bb, acc[j]);
        }
    }

    // den lives in tile 4, column 64 -> lanes with (lane&15)==0
    if ((lane & 15) == 0) {
#pragma unroll
        for (int vv = 0; vv < 8; vv++) {
            int rr = wave * 16 + vv + ((lane & 16) ? 8 : 0);
            den_s[rr] = acc[4][vv];
        }
    }
    __syncthreads();

#pragma unroll
    for (int j = 0; j < 4; j++) {
        int d = j * 16 + (lane & 15);
#pragma unroll
        for (int vv = 0; vv < 8; vv++) {
            int rloc = wave * 16 + vv + ((lane & 16) ? 8 : 0);
            float dv = den_s[rloc] + 1e-6f;
            out[((size_t)(b * SS + s0 + rloc)) * (HH * DVV) + h * DVV + d] =
                acc[j][vv] / dv;
        }
    }
}

// ---------------------------------------------------------------------------
// Launch
// ---------------------------------------------------------------------------
extern "C" void kernel_launch(void* const* d_in, const int* in_sizes, int n_in,
                              void* d_out, int out_size, void* d_ws, size_t ws_size,
                              hipStream_t stream) {
    const float* x    = (const float*)d_in[0];
    const float* ti   = (const float*)d_in[1];
    const float* Wq   = (const float*)d_in[2];
    const float* bq   = (const float*)d_in[3];
    const float* Wk   = (const float*)d_in[4];
    const float* bk   = (const float*)d_in[5];
    const float* Wv   = (const float*)d_in[6];
    const float* bv   = (const float*)d_in[7];
    const float* Wphi = (const float*)d_in[8];
    const float* Wt   = (const float*)d_in[9];
    const float* bt   = (const float*)d_in[10];
    float* out = (float*)d_out;

    char* ws = (char*)d_ws;
    size_t off = 0;
    auto alloc = [&](size_t bytes) -> void* {
        void* p = ws + off;
        off = (off + bytes + 255) & ~(size_t)255;
        return p;
    };

    const size_t NBIG = (size_t)MROWS * 1024;      // 33.5M elems
    unsigned short* x_bf    = (unsigned short*)alloc(NBIG * 2);
    unsigned short* q_bf    = (unsigned short*)alloc(NBIG * 2);  // later reused for pk
    unsigned short* k_bf    = (unsigned short*)alloc(NBIG * 2);
    unsigned short* v_bf    = (unsigned short*)alloc(NBIG * 2);
    unsigned short* phiq_bf = (unsigned short*)alloc(NBIG * 2);
    unsigned short* WqT     = (unsigned short*)alloc((size_t)EE * 1024 * 2);
    unsigned short* WkT     = (unsigned short*)alloc((size_t)EE * 1024 * 2);
    unsigned short* WvT     = (unsigned short*)alloc((size_t)EE * 1024 * 2);
    unsigned short* WphiT   = (unsigned short*)alloc((size_t)FF * 1024 * 2);
    float*          Tarr    = (float*)alloc((size_t)MROWS * HH * 4);
    float*          kvf     = (float*)alloc((size_t)BB * HH * 64 * 80 * 4);
    unsigned short* kvT     = (unsigned short*)alloc((size_t)BB * HH * 96 * 64 * 2);
    unsigned short* pk_bf   = q_bf;   // q is dead once phi_q is computed

    // 1) conversions: x elementwise, weights transposed to (N,K)
    cvt_bf16_kernel<<<8192, 256, 0, stream>>>(x, x_bf, NBIG);
    dim3 tgrid(1024 / 32, 1024 / 32);
    transpose_cvt_kernel<<<tgrid, 256, 0, stream>>>(Wq, WqT, 1024, 1024);
    transpose_cvt_kernel<<<tgrid, 256, 0, stream>>>(Wk, WkT, 1024, 1024);
    transpose_cvt_kernel<<<tgrid, 256, 0, stream>>>(Wv, WvT, 1024, 1024);
    transpose_cvt_kernel<<<tgrid, 256, 0, stream>>>(Wphi, WphiT, 1024, 1024);

    // 2) temperatures
    temp_kernel<<<MROWS / 16, 256, 0, stream>>>(ti, Wt, bt, Tarr);

    // 3) projections (WMMA GEMMs, async-LDS staged)
    dim3 ggrid(1024 / 128, MROWS / 128);   // (8, 256)
    gemm_bias_act_kernel<<<ggrid, 256, 0, stream>>>(x_bf, WqT, bq, nullptr, q_bf,
                                                    MROWS, 1024, 1024, 0);
    gemm_bias_act_kernel<<<ggrid, 256, 0, stream>>>(x_bf, WkT, bk, nullptr, k_bf,
                                                    MROWS, 1024, 1024, 0);
    gemm_bias_act_kernel<<<ggrid, 256, 0, stream>>>(x_bf, WvT, bv, nullptr, v_bf,
                                                    MROWS, 1024, 1024, 0);
    // 4) feature maps
    gemm_bias_act_kernel<<<ggrid, 256, 0, stream>>>(q_bf, WphiT, nullptr, nullptr,
                                                    phiq_bf, MROWS, 1024, 1024, 1);
    gemm_bias_act_kernel<<<ggrid, 256, 0, stream>>>(k_bf, WphiT, nullptr, Tarr,
                                                    pk_bf, MROWS, 1024, 1024, 2);

    // 5) kv (+ fused z in augmented column 64)
    zero_f32_kernel<<<512, 256, 0, stream>>>(kvf, (size_t)BB * HH * 64 * 80);
    kv_kernel<<<dim3(BB * HH, KSLICES), 256, 0, stream>>>(pk_bf, v_bf, kvf);
    cvt_kv_kernel<<<1536, 256, 0, stream>>>(kvf, kvT);

    // 6) numerator / denominator / divide
    out_kernel<<<dim3(BB * HH, SS / 128), 256, 0, stream>>>(phiq_bf, kvT, out);
}
